// RelMultiHeadedAttention_31095563223737
// MI455X (gfx1250) — matrix operand order repaired and checked by reading, not compiled, not run on hardware
//
#include <hip/hip_runtime.h>

// ---------------- types & helpers ----------------
typedef __attribute__((ext_vector_type(16))) __bf16 v16bf;
typedef __attribute__((ext_vector_type(8)))  float  v8f;
typedef __attribute__((ext_vector_type(4)))  unsigned int v4u;
typedef __attribute__((ext_vector_type(4)))  unsigned int u32x4;
typedef __attribute__((ext_vector_type(8)))  int i32x8;
typedef __attribute__((ext_vector_type(4)))  int i32x4;

// TDM builtin: 5-arg form on ROCm 7.2 (clang-22); guard so other toolchains
// (6-arg clang-23 headers, host pass) fall back to the manual copy path.
#if defined(__AMDGCN__) && defined(__gfx1250__) && \
    __has_builtin(__builtin_amdgcn_tensor_load_to_lds) && (__clang_major__ < 23)
#define USE_TDM 1
#else
#define USE_TDM 0
#endif

union Frag { v16bf bf; v4u q[2]; };

__device__ __forceinline__ unsigned short f2bf(float f) {
  unsigned int x = __builtin_bit_cast(unsigned int, f);
  unsigned int r = x + 0x7FFFu + ((x >> 16) & 1u);   // RNE
  return (unsigned short)(r >> 16);
}
__device__ __forceinline__ float bf2f(unsigned short u) {
  unsigned int x = ((unsigned int)u) << 16;
  return __builtin_bit_cast(float, x);
}

// Async global -> LDS 16B copy (ASYNCcnt-tracked, no VGPR round trip).
__device__ __forceinline__ void async_cp16(const void* gsrc, void* lds_dst) {
  unsigned ldsoff = (unsigned)(size_t)lds_dst;          // low 32 bits = LDS byte offset
  unsigned long long ga = (unsigned long long)gsrc;
  asm volatile("global_load_async_to_lds_b128 %0, %1, off"
               :: "v"(ldsoff), "v"(ga) : "memory");
}
// Wait until <= n async ops outstanding (in-order completion per wave).
#define ASYNC_WAIT(n) asm volatile("s_wait_asynccnt " #n ::: "memory")

// A fragment (16x32 bf16): lane m=lane&15, half h=lane>>4.
// elements 0..7  <-> K = ks + 8h + 0..7 ; elements 8..15 <-> K = ks + 16 + 8h + 0..7
__device__ __forceinline__ void load_a(Frag& f, const unsigned short* lds,
                                       int rowBase, int ld, int ks, int lane) {
  const int m = lane & 15, h = lane >> 4;
  const unsigned short* p = lds + (rowBase + m) * ld + ks + 8 * h;
  f.q[0] = *(const v4u*)p;
  f.q[1] = *(const v4u*)(p + 16);
}
// B fragment (32x16 bf16, K x N): lane n=lane&15, half h=lane>>4.
// elements 0..7  <-> K = ks + 8h + 0..7 ; elements 8..15 <-> K = ks + 16 + 8h + 0..7
__device__ __forceinline__ void load_b(Frag& f, const unsigned short* lds,
                                       int nBase, int ld, int ks, int lane) {
  const int n = lane & 15, h = lane >> 4;
  const unsigned short* p = lds + (nBase + n) * ld + ks + 8 * h;
  f.q[0] = *(const v4u*)p;
  f.q[1] = *(const v4u*)(p + 16);
}
__device__ __forceinline__ v8f wmma_bf16(const Frag& a, const Frag& b, v8f c) {
  return __builtin_amdgcn_wmma_f32_16x16x32_bf16(false, a.bf, false, b.bf,
                                                 (short)0, c, false, false);
}

// ---------------- fp32 -> bf16 convert ----------------
__global__ void cvt_f32_bf16(const float* __restrict__ in,
                             unsigned short* __restrict__ out, int n) {
  int i = blockIdx.x * 256 + threadIdx.x;
  if (i < n) out[i] = f2bf(in[i]);
}

// ---------------- generic batched GEMM: C = A(MxK) * Bw(NxK)^T ----------------
// Double-buffered async-to-LDS pipeline: issue tile i+1 while computing tile i.
// batch index z: zb=z/nH, zh=z%nH ; pointer offsets via (sXb, sXh).
// Outputs: C0/C1 bf16 (each with up to 2 fused bias vectors), Cf f32 (bias bfa).
__global__ __launch_bounds__(256)
void gemm_bf16(const unsigned short* __restrict__ A, int lda, long long sAb, long long sAh,
               const unsigned short* __restrict__ Bw, int ldb, long long sBb, long long sBh,
               int M, int N, int K, int nH,
               unsigned short* __restrict__ C0, const float* b0a, const float* b0b,
               unsigned short* __restrict__ C1, const float* b1a, const float* b1b,
               float* __restrict__ Cf, const float* bfa,
               int ldc, long long sCb, long long sCh) {
  __shared__ unsigned short sA[2][128 * 32];
  __shared__ unsigned short sB[2][64 * 32];

  const int z = blockIdx.z, zb = z / nH, zh = z - zb * nH;
  const unsigned short* Ab = A + (size_t)zb * sAb + (size_t)zh * sAh;
  const unsigned short* Bb = Bw + (size_t)zb * sBb + (size_t)zh * sBh;
  const long long cOff = (long long)zb * sCb + (long long)zh * sCh;

  const int m0 = blockIdx.y * 128, n0 = blockIdx.x * 64;
  const int tid = threadIdx.x, wave = tid >> 5, lane = tid & 31;
  const int nl = lane & 15, half = lane >> 4;
  const int wm = wave >> 1, wn = wave & 1;   // 4x2 wave grid, 32x32 per wave

  // 3 async 16B issues per thread per tile (A: 2, B: 1)
  auto issue_tile = [&](int k0, int buf) {
#pragma unroll
    for (int it = 0; it < 2; it++) {
      int i = tid + it * 256;
      int r = i >> 2, cv = i & 3;
      async_cp16(&Ab[(size_t)(m0 + r) * lda + k0 + cv * 8], &sA[buf][r * 32 + cv * 8]);
    }
    int r = tid >> 2, cv = tid & 3;
    async_cp16(&Bb[(size_t)(n0 + r) * ldb + k0 + cv * 8], &sB[buf][r * 32 + cv * 8]);
  };

  v8f acc[2][2];
#pragma unroll
  for (int i = 0; i < 2; i++)
#pragma unroll
    for (int j = 0; j < 2; j++)
#pragma unroll
      for (int e = 0; e < 8; e++) acc[i][j][e] = 0.f;

  const int nIter = K >> 5;
  issue_tile(0, 0);
  for (int it = 0; it < nIter; ++it) {
    const int cur = it & 1;
    __syncthreads();                       // prior readers of buf[1-cur] are done
    if (it + 1 < nIter) {
      issue_tile((it + 1) << 5, 1 - cur);  // prefetch next tile into other buffer
      __builtin_prefetch(&Ab[(size_t)(m0 + (tid >> 2)) * lda + ((it + 2) << 5)], 0, 0);
      ASYNC_WAIT(0x3);                     // newest 3 (next tile) may stay in flight
    } else {
      ASYNC_WAIT(0x0);
    }
    __syncthreads();                       // publish buf[cur] to all waves

#pragma unroll
    for (int i = 0; i < 2; i++) {
      Frag a; load_a(a, sA[cur], wm * 32 + i * 16, 32, 0, lane);
#pragma unroll
      for (int j = 0; j < 2; j++) {
        Frag bq; load_b(bq, sB[cur], wn * 32 + j * 16, 32, 0, lane);
        acc[i][j] = wmma_bf16(a, bq, acc[i][j]);
      }
    }
  }

#pragma unroll
  for (int i = 0; i < 2; i++)
#pragma unroll
    for (int j = 0; j < 2; j++)
#pragma unroll
      for (int r = 0; r < 8; r++) {
        const int row = m0 + wm * 32 + i * 16 + r + 8 * half;
        const int col = n0 + wn * 32 + j * 16 + nl;
        const float v = acc[i][j][r];
        const size_t idx = (size_t)(cOff + (long long)row * ldc + col);
        if (C0) {
          float bs = (b0a ? b0a[col] : 0.f) + (b0b ? b0b[col] : 0.f);
          C0[idx] = f2bf(v + bs);
        }
        if (C1) {
          float bs = (b1a ? b1a[col] : 0.f) + (b1b ? b1b[col] : 0.f);
          C1[idx] = f2bf(v + bs);
        }
        if (Cf) Cf[idx] = v + (bfa ? bfa[col] : 0.f);
      }
}

// ---------------- fused rel-shift flash attention ----------------
// Qu,Kh,Vh : [B,T,D] bf16 ; G : [B*H,T,T] bf16 (unshifted Qv·Ph^T) ;
// mask : [B,T,T] i32 ; Out : [B,T,D] bf16
__global__ __launch_bounds__(256)
void flash_relattn(const unsigned short* __restrict__ Qu,
                   const unsigned short* __restrict__ Kh,
                   const unsigned short* __restrict__ Vh,
                   const unsigned short* __restrict__ G,
                   const int* __restrict__ mask,
                   unsigned short* __restrict__ Out) {
  constexpr int T = 1024, D = 512, H = 8;
  __shared__ unsigned short sQ[128 * 64];
  __shared__ unsigned short sK[64 * 64];
  __shared__ unsigned short sV[64 * 72];   // transposed [d][p], padded
  __shared__ unsigned short sP[128 * 64];  // wave-private strips

  const int b = blockIdx.z, h = blockIdx.y;
  const int u0 = blockIdx.x * 128;
  const int tid = threadIdx.x, wave = tid >> 5, lane = tid & 31;
  const int nl = lane & 15, half = lane >> 4;
  const float NEG = -3.0e38f;

  { // Q block 128x64 via async-to-LDS
    const unsigned short* src = Qu + (size_t)(b * T + u0) * D + h * 64;
#pragma unroll
    for (int it = 0; it < 4; it++) {
      int i = tid + it * 256;
      int r = i >> 3, cv = i & 7;
      async_cp16(&src[(size_t)r * D + cv * 8], &sQ[r * 64 + cv * 8]);
    }
    ASYNC_WAIT(0x0);
  }

  v8f O[4];
  float rm[8], rl[8];
#pragma unroll
  for (int j = 0; j < 4; j++)
#pragma unroll
    for (int e = 0; e < 8; e++) O[j][e] = 0.f;
#pragma unroll
  for (int r = 0; r < 8; r++) { rm[r] = NEG; rl[r] = 0.f; }

  for (int p0 = 0; p0 < T; p0 += 64) {
    __syncthreads();
    const unsigned short* ksrc = Kh + (size_t)(b * T + p0) * D + h * 64;
#if USE_TDM
    if (wave == 0) {
      // Tensor Data Mover: 64x64 bf16 tile, row stride D elems -> contiguous sK
      const unsigned long long ga = (unsigned long long)(const void*)ksrc;
      const unsigned lk = (unsigned)(size_t)(void*)sK;
      u32x4 g0 = { 1u,                                   // count=1, user mode
                   lk,                                   // lds_addr (bytes)
                   (unsigned)ga,                         // global_addr[31:0]
                   ((unsigned)((ga >> 32) & 0x01FFFFFFull)) | 0x80000000u }; // addr[56:32] | type=2
      i32x8 g1 = { 0x00010000,      // data_size=1 (2 bytes), mask=0
                   64 << 16,        // tensor_dim0 = 64 (low16 in bits 63:48)
                   64 << 16,        // tensor_dim1 = 64 (low16 in bits 95:80)
                   64 << 16,        // tile_dim0   = 64 (bits 127:112)
                   64,              // tile_dim1   = 64 (bits 143:128), tile_dim2 = 0
                   512,             // tensor_dim0_stride = 512 elems (bits 191:160)
                   0, 0 };
      i32x4 gz = { 0, 0, 0, 0 };
      __builtin_amdgcn_tensor_load_to_lds(g0, g1, gz, gz, 0);
    }
#else
    { // K tile 64x64 -> sK[p][d]
#pragma unroll
      for (int it = 0; it < 2; it++) {
        int i = tid + it * 256;
        int r = i >> 3, cv = i & 7;
        *(v4u*)&sK[r * 64 + cv * 8] = *(const v4u*)&ksrc[(size_t)r * D + cv * 8];
      }
    }
#endif
    { // V tile 64x64 -> sV[d][p] (transpose)
      const unsigned short* vsrc = Vh + (size_t)(b * T + p0) * D + h * 64;
#pragma unroll
      for (int it = 0; it < 2; it++) {
        int i = tid + it * 256;
        int r = i >> 3, cv = i & 7;
        v4u vv = *(const v4u*)&vsrc[(size_t)r * D + cv * 8];
        const unsigned short* e = (const unsigned short*)&vv;
#pragma unroll
        for (int t2 = 0; t2 < 8; t2++) sV[(cv * 8 + t2) * 72 + r] = e[t2];
      }
    }
#if USE_TDM
    if (wave == 0) __builtin_amdgcn_s_wait_tensorcnt(0);
#endif
    __syncthreads();

    // ---- AC = Qu * K^T ----
    v8f S[4];
#pragma unroll
    for (int j = 0; j < 4; j++)
#pragma unroll
      for (int e = 0; e < 8; e++) S[j][e] = 0.f;
#pragma unroll
    for (int ks = 0; ks < 64; ks += 32) {
      Frag a; load_a(a, sQ, wave * 16, 64, ks, lane);
#pragma unroll
      for (int j = 0; j < 4; j++) {
        Frag bq; load_b(bq, sK, j * 16, 64, ks, lane);
        S[j] = wmma_bf16(a, bq, S[j]);
      }
    }

    // ---- add shifted BD (gather from G), mask, scale ----
    float mloc[8];
#pragma unroll
    for (int r = 0; r < 8; r++) mloc[r] = NEG;
#pragma unroll
    for (int j = 0; j < 4; j++)
#pragma unroll
      for (int r = 0; r < 8; r++) {
        const int u = u0 + wave * 16 + r + 8 * half;
        const int p = p0 + j * 16 + nl;
        float g = 0.f;
        if (p != u + 1) {
          int gr, gc;
          if (p <= u) { gr = u;     gc = p - u + (T - 1); }
          else        { gr = u + 1; gc = p - u - 2;       }
          g = bf2f(G[((size_t)(b * H + h) * T + gr) * T + gc]);
        }
        const int mv = mask[(size_t)(b * T + u) * T + p];
        float s = (mv == 0) ? NEG : (S[j][r] + g) * 0.125f;
        S[j][r] = s;
        mloc[r] = fmaxf(mloc[r], s);
      }

    // ---- online softmax (rows live in 16-lane halves) ----
#pragma unroll
    for (int r = 0; r < 8; r++) {
#pragma unroll
      for (int off = 1; off < 16; off <<= 1)
        mloc[r] = fmaxf(mloc[r], __shfl_xor(mloc[r], off, 32));
      const float mn = fmaxf(rm[r], mloc[r]);
      const float sc = __expf(rm[r] - mn);
      rm[r] = mn;
      rl[r] *= sc;
#pragma unroll
      for (int j = 0; j < 4; j++) O[j][r] *= sc;
    }
    float psum[8];
#pragma unroll
    for (int r = 0; r < 8; r++) psum[r] = 0.f;
#pragma unroll
    for (int j = 0; j < 4; j++)
#pragma unroll
      for (int r = 0; r < 8; r++) {
        const float pe = __expf(S[j][r] - rm[r]);
        psum[r] += pe;
        sP[(wave * 16 + r + 8 * half) * 64 + j * 16 + nl] = f2bf(pe);
      }
#pragma unroll
    for (int r = 0; r < 8; r++) {
#pragma unroll
      for (int off = 1; off < 16; off <<= 1)
        psum[r] += __shfl_xor(psum[r], off, 32);
      rl[r] += psum[r];
    }

    // ---- O += P * V ----
#pragma unroll
    for (int ks = 0; ks < 64; ks += 32) {
      Frag a; load_a(a, sP, wave * 16, 64, ks, lane);
#pragma unroll
      for (int j = 0; j < 4; j++) {
        Frag bv; load_b(bv, sV, j * 16, 72, ks, lane);
        O[j] = wmma_bf16(a, bv, O[j]);
      }
    }
  }

  // ---- epilogue ----
#pragma unroll
  for (int j = 0; j < 4; j++)
#pragma unroll
    for (int r = 0; r < 8; r++) {
      const int u = u0 + wave * 16 + r + 8 * half;
      const float inv = (rl[r] > 0.f) ? 1.f / rl[r] : 0.f;
      Out[(size_t)(b * T + u) * D + h * 64 + j * 16 + nl] = f2bf(O[j][r] * inv);
    }
}

// ---------------- host launch ----------------
extern "C" void kernel_launch(void* const* d_in, const int* in_sizes, int n_in,
                              void* d_out, int out_size, void* d_ws, size_t ws_size,
                              hipStream_t stream) {
  constexpr int Bn = 8, T = 1024, D = 512, H = 8;
  const float* q    = (const float*)d_in[0];
  const float* k    = (const float*)d_in[1];
  const float* v    = (const float*)d_in[2];
  const float* pos  = (const float*)d_in[3];
  const int*   mask = (const int*)d_in[4];
  const float* Wq = (const float*)d_in[5];
  const float* bq = (const float*)d_in[6];
  const float* Wk = (const float*)d_in[7];
  const float* bk = (const float*)d_in[8];
  const float* Wv = (const float*)d_in[9];
  const float* bv = (const float*)d_in[10];
  const float* Wo = (const float*)d_in[11];
  const float* bo = (const float*)d_in[12];
  const float* Wp = (const float*)d_in[13];
  const float* bias_u = (const float*)d_in[14];
  const float* bias_v = (const float*)d_in[15];

  char* ws = (char*)d_ws;
  size_t off = 0;
  auto alloc = [&](size_t bytes) {
    void* p = ws + off;
    off = (off + bytes + 255) & ~(size_t)255;
    return p;
  };
  unsigned short* wq16 = (unsigned short*)alloc((size_t)D * D * 2);
  unsigned short* wk16 = (unsigned short*)alloc((size_t)D * D * 2);
  unsigned short* wv16 = (unsigned short*)alloc((size_t)D * D * 2);
  unsigned short* wp16 = (unsigned short*)alloc((size_t)D * D * 2);
  unsigned short* wo16 = (unsigned short*)alloc((size_t)D * D * 2);
  unsigned short* q16  = (unsigned short*)alloc((size_t)Bn * T * D * 2);
  unsigned short* k16  = (unsigned short*)alloc((size_t)Bn * T * D * 2);
  unsigned short* v16  = (unsigned short*)alloc((size_t)Bn * T * D * 2);
  unsigned short* pos16= (unsigned short*)alloc((size_t)T * D * 2);
  unsigned short* qu16 = (unsigned short*)alloc((size_t)Bn * T * D * 2);
  unsigned short* qv16 = (unsigned short*)alloc((size_t)Bn * T * D * 2);
  unsigned short* kh16 = (unsigned short*)alloc((size_t)Bn * T * D * 2);
  unsigned short* vh16 = (unsigned short*)alloc((size_t)Bn * T * D * 2);
  unsigned short* ph16 = (unsigned short*)alloc((size_t)T * D * 2);
  unsigned short* at16 = (unsigned short*)alloc((size_t)Bn * T * D * 2);
  unsigned short* G16  = (unsigned short*)alloc((size_t)Bn * H * T * T * 2);

  auto cvt = [&](const float* src, unsigned short* dst, int n) {
    cvt_f32_bf16<<<dim3((n + 255) / 256), dim3(256), 0, stream>>>(src, dst, n);
  };
  cvt(q, q16, Bn * T * D);
  cvt(k, k16, Bn * T * D);
  cvt(v, v16, Bn * T * D);
  cvt(pos, pos16, T * D);
  cvt(Wq, wq16, D * D);
  cvt(Wk, wk16, D * D);
  cvt(Wv, wv16, D * D);
  cvt(Wp, wp16, D * D);
  cvt(Wo, wo16, D * D);

  const dim3 blk(256);
  // qh -> qu (bq + bias_u), qv (bq + bias_v)
  gemm_bf16<<<dim3(D / 64, Bn * T / 128, 1), blk, 0, stream>>>(
      q16, D, 0, 0, wq16, D, 0, 0, Bn * T, D, D, 1,
      qu16, bq, bias_u, qv16, bq, bias_v, nullptr, nullptr, D, 0, 0);
  // kh
  gemm_bf16<<<dim3(D / 64, Bn * T / 128, 1), blk, 0, stream>>>(
      k16, D, 0, 0, wk16, D, 0, 0, Bn * T, D, D, 1,
      kh16, bk, nullptr, nullptr, nullptr, nullptr, nullptr, nullptr, D, 0, 0);
  // vh
  gemm_bf16<<<dim3(D / 64, Bn * T / 128, 1), blk, 0, stream>>>(
      v16, D, 0, 0, wv16, D, 0, 0, Bn * T, D, D, 1,
      vh16, bv, nullptr, nullptr, nullptr, nullptr, nullptr, nullptr, D, 0, 0);
  // ph (no bias)
  gemm_bf16<<<dim3(D / 64, T / 128, 1), blk, 0, stream>>>(
      pos16, D, 0, 0, wp16, D, 0, 0, T, D, D, 1,
      ph16, nullptr, nullptr, nullptr, nullptr, nullptr, nullptr, nullptr, D, 0, 0);
  // G[b,h] = Qv_bh (TxdK) * Ph_h(TxdK)^T, batched over 64 (b,h)
  gemm_bf16<<<dim3(T / 64, T / 128, Bn * H), blk, 0, stream>>>(
      qv16, D, (long long)T * D, 64, ph16, D, 0, 64, T, T, 64, H,
      G16, nullptr, nullptr, nullptr, nullptr, nullptr, nullptr, nullptr,
      T, (long long)H * T * T, (long long)T * T);
  // fused rel-shift flash attention
  flash_relattn<<<dim3(T / 128, H, Bn), blk, 0, stream>>>(
      qu16, kh16, vh16, G16, mask, at16);
  // final projection -> f32 out
  gemm_bf16<<<dim3(D / 64, Bn * T / 128, 1), blk, 0, stream>>>(
      at16, D, 0, 0, wo16, D, 0, 0, Bn * T, D, D, 1,
      nullptr, nullptr, nullptr, nullptr, nullptr, nullptr,
      (float*)d_out, bo, D, 0, 0);
}
